// Sequence_74380243632219
// MI455X (gfx1250) — matrix operand order repaired and checked by reading, not compiled, not run on hardware
//
#include <hip/hip_runtime.h>
#include <hip/hip_bf16.h>
#include <math.h>

// ---- problem sizes (fixed by the reference) ----
#define TT  512
#define BB  256
#define INN 8
#define HH  512
#define OUTN 5

typedef __attribute__((ext_vector_type(16))) __bf16 v16bf;
typedef __attribute__((ext_vector_type(8)))  __bf16 v8bf;
typedef __attribute__((ext_vector_type(8)))  float  v8f;

__device__ __forceinline__ __bf16 f2bf(float f) {
    union { float f; unsigned u; } c; c.f = f;
    unsigned r = c.u + 0x7FFFu + ((c.u >> 16) & 1u);   // round-to-nearest-even
    unsigned short h = (unsigned short)(r >> 16);
    __bf16 b; __builtin_memcpy(&b, &h, 2); return b;
}

__device__ __forceinline__ float sigmoidf_(float x) {
    return 1.0f / (1.0f + __expf(-x));
}

// ---------------- CDNA5 async global->LDS staging ----------------
#if __has_builtin(__builtin_amdgcn_global_load_async_to_lds_b128)
#define HAVE_ASYNC_LDS 1
#endif

typedef int b128i __attribute__((vector_size(16)));                 // 16-byte payload
typedef __attribute__((address_space(1))) b128i* g128p;             // global ptr
typedef __attribute__((address_space(3))) b128i* l128p;             // LDS ptr

__device__ __forceinline__ void async_cp16(__bf16* lds_dst, const __bf16* gsrc) {
#ifdef HAVE_ASYNC_LDS
    // generic shared ptr: low 32 bits are the LDS byte offset (ISA flat->LDS rule)
    g128p g = (g128p)(uintptr_t)(const void*)gsrc;
    l128p l = (l128p)(unsigned)(uintptr_t)(void*)lds_dst;
    __builtin_amdgcn_global_load_async_to_lds_b128(g, l, 0, 0);
#else
    *(v8bf*)lds_dst = *(const v8bf*)gsrc;
#endif
}

__device__ __forceinline__ void async_wait() {
#ifdef HAVE_ASYNC_LDS
#if __has_builtin(__builtin_amdgcn_s_wait_asynccnt)
    __builtin_amdgcn_s_wait_asynccnt(0);
#else
    asm volatile("s_wait_asynccnt 0x0" ::: "memory");
#endif
#endif
}

// A fragment (16x32 bf16, M x K) per ISA layout:
//   m = lane&15, half = lane>>4
//   elements 0..7  = A[m][k0 + half*8 .. +8)
//   elements 8..15 = A[m][k0 + 16 + half*8 .. +8)
__device__ __forceinline__ v16bf load_a_frag(const __bf16* base, int row0, int k0, int lane) {
    int m = lane & 15, half = lane >> 4;
    const __bf16* p = base + (size_t)(row0 + m) * HH + k0 + half * 8;
    v8bf lo = *(const v8bf*)(p);
    v8bf hi = *(const v8bf*)(p + 16);
    return __builtin_shufflevector(lo, hi, 0,1,2,3,4,5,6,7,8,9,10,11,12,13,14,15);
}

// B fragment (32x16 bf16, K x N) stored in LDS, 32 contiguous bytes per lane
__device__ __forceinline__ v16bf lds_frag(const __bf16* lds, int slot, int lane) {
    return *(const v16bf*)(lds + ((size_t)slot * 32 + lane) * 16);
}

#define WMMA_BF16(A, Bm, C) \
    __builtin_amdgcn_wmma_f32_16x16x32_bf16(false, (A), false, (Bm), (short)0, (C), false, false)

// ---------------- weight pre-pack: fp32 [3H][H] -> bf16 LDS fragment image ----
// dst layout: [32 jb][16 ks][3 g][32 lane][16 elems]  (48 KB contiguous per jb)
__global__ void gru_pack_kernel(const float* __restrict__ W, __bf16* __restrict__ dst) {
    int fl = blockIdx.x * blockDim.x + threadIdx.x;   // fragment-lane id
    if (fl >= 32 * 16 * 3 * 32) return;
    int lane = fl & 31;
    int g    = (fl >> 5) % 3;
    int ks   = (fl / 96) & 15;
    int jb   = fl / 1536;
    int n = lane & 15, half = lane >> 4;
    const float* src = W + (size_t)(g * HH + jb * 16 + n) * HH + ks * 32 + half * 8;
    __bf16* d = dst + (size_t)fl * 16;
#pragma unroll
    for (int i = 0; i < 16; ++i) d[i] = f2bf(src[(i & 7) + (i >> 3) * 16]);
}

// ---------------- init: zero parity-0 h buffers ----------------
__global__ void gru_init_kernel(float* h0f, float* h1f, __bf16* h0b, __bf16* h1b) {
    int i = blockIdx.x * blockDim.x + threadIdx.x;
    if (i >= BB * HH) return;
    h0f[i] = 0.0f; h1f[i] = 0.0f;
    __bf16 z = f2bf(0.0f);
    h0b[i] = z;    h1b[i] = z;
}

// ---------------- layer 0: h0 = GRUCell(x_t, h0) ----------------
// grid = H/16 j-blocks, block = 256 (8 waves); wave handles 32 batch rows (2 tiles)
__global__ __launch_bounds__(256)
void gru_layer0_kernel(const float* __restrict__ x, int t,
                       const float* __restrict__ W_ih0,
                       const __bf16* __restrict__ packW0,   // prepacked W_hh0 image
                       const float* __restrict__ b_ih0, const float* __restrict__ b_hh0,
                       const float*  __restrict__ h0f_in, const __bf16* __restrict__ h0b_in,
                       float* __restrict__ h0f_out, __bf16* __restrict__ h0b_out) {
    __shared__ __align__(32) __bf16 ldsW[16 * 3 * 32 * 16];   // 48 KB

    const int jb = blockIdx.x;
    const int j0 = jb * 16;

    // stage the prepacked 48 KB image: pure linear async copy
    {
        const __bf16* src = packW0 + (size_t)jb * (16 * 3 * 32 * 16);
        for (int u = threadIdx.x; u < 3072; u += blockDim.x)   // 3072 x 16B
            async_cp16(&ldsW[u * 8], src + u * 8);
        async_wait();
        __syncthreads();
    }

    const int lane = threadIdx.x & 31;
    const int wave = threadIdx.x >> 5;
    const int b0   = wave * 32;          // two 16-row tiles per wave

    v8f accR[2] = {{}, {}}, accZ[2] = {{}, {}}, accN[2] = {{}, {}};
#pragma unroll 4
    for (int ks = 0; ks < 16; ++ks) {
        v16bf a0 = load_a_frag(h0b_in, b0,      ks * 32, lane);
        v16bf a1 = load_a_frag(h0b_in, b0 + 16, ks * 32, lane);
        v16bf bR = lds_frag(ldsW, ks * 3 + 0, lane);
        v16bf bZ = lds_frag(ldsW, ks * 3 + 1, lane);
        v16bf bN = lds_frag(ldsW, ks * 3 + 2, lane);
        accR[0] = WMMA_BF16(a0, bR, accR[0]);
        accR[1] = WMMA_BF16(a1, bR, accR[1]);
        accZ[0] = WMMA_BF16(a0, bZ, accZ[0]);
        accZ[1] = WMMA_BF16(a1, bZ, accZ[1]);
        accN[0] = WMMA_BF16(a0, bN, accN[0]);
        accN[1] = WMMA_BF16(a1, bN, accN[1]);
    }

    // gate math; C/D layout: n = lane&15, m = vgpr_i + (lane>=16 ? 8 : 0)
    const int n = lane & 15, half = lane >> 4;
    const int j = j0 + n;
    const float bihr = b_ih0[j], bihz = b_ih0[HH + j], bihn = b_ih0[2 * HH + j];
    const float bhhr = b_hh0[j], bhhz = b_hh0[HH + j], bhhn = b_hh0[2 * HH + j];
    float wiR[INN], wiZ[INN], wiN[INN];
#pragma unroll
    for (int k = 0; k < INN; ++k) {
        wiR[k] = W_ih0[(size_t)(0 * HH + j) * INN + k];
        wiZ[k] = W_ih0[(size_t)(1 * HH + j) * INN + k];
        wiN[k] = W_ih0[(size_t)(2 * HH + j) * INN + k];
    }
    const float* xt = x + (size_t)t * BB * INN;

#pragma unroll
    for (int tt = 0; tt < 2; ++tt) {
#pragma unroll
        for (int i = 0; i < 8; ++i) {
            int b = b0 + tt * 16 + half * 8 + i;
            const float* xb = xt + (size_t)b * INN;
            float gr = bihr, gz = bihz, gn = bihn;
#pragma unroll
            for (int k = 0; k < INN; ++k) {
                gr += xb[k] * wiR[k];
                gz += xb[k] * wiZ[k];
                gn += xb[k] * wiN[k];
            }
            float r  = sigmoidf_(gr + accR[tt][i] + bhhr);
            float z  = sigmoidf_(gz + accZ[tt][i] + bhhz);
            float ng = tanhf(gn + r * (accN[tt][i] + bhhn));
            float hp = h0f_in[(size_t)b * HH + j];
            float hn = (1.0f - z) * ng + z * hp;
            h0f_out[(size_t)b * HH + j] = hn;
            h0b_out[(size_t)b * HH + j] = f2bf(hn);
        }
    }
}

// ---------------- layer 1: h1 = GRUCell(h0_new, h1) ----------------
// LDS layout per chunk: [8 ksl][6 gg][32 lane][16]; gg 0..2 = W_ih1, 3..5 = W_hh1
__global__ __launch_bounds__(256)
void gru_layer1_kernel(const __bf16* __restrict__ packIH1, const __bf16* __restrict__ packHH1,
                       const float* __restrict__ b_ih1, const float* __restrict__ b_hh1,
                       const __bf16* __restrict__ h0b_cur,
                       const float*  __restrict__ h1f_in, const __bf16* __restrict__ h1b_in,
                       float* __restrict__ h1f_out, __bf16* __restrict__ h1b_out) {
    __shared__ __align__(32) __bf16 ldsW[8 * 6 * 32 * 16];   // 48 KB

    const int jb   = blockIdx.x;
    const int j0   = jb * 16;
    const int lane = threadIdx.x & 31;
    const int wave = threadIdx.x >> 5;
    const int b0   = wave * 32;

    v8f aR[2] = {{}, {}}, aZ[2] = {{}, {}}, aN[2] = {{}, {}};   // input-side
    v8f hR[2] = {{}, {}}, hZ[2] = {{}, {}}, hN[2] = {{}, {}};   // hidden-side

    for (int ck = 0; ck < 2; ++ck) {
        if (ck) __syncthreads();     // previous chunk fully consumed
        {   // each ks image is 3 KB (192 x 16B) per matrix; interleave IH/HH per ksl
            const __bf16* sIH = packIH1 + (size_t)jb * 24576 + (size_t)ck * 12288;
            const __bf16* sHH = packHH1 + (size_t)jb * 24576 + (size_t)ck * 12288;
            for (int u = threadIdx.x; u < 1536; u += blockDim.x) {
                int ksl = u / 192, w = u - ksl * 192;
                async_cp16(&ldsW[(ksl * 384 + w) * 8],       sIH + (size_t)u * 8);
                async_cp16(&ldsW[(ksl * 384 + 192 + w) * 8], sHH + (size_t)u * 8);
            }
            async_wait();
            __syncthreads();
        }

#pragma unroll 2
        for (int ksl = 0; ksl < 8; ++ksl) {
            int ks = ck * 8 + ksl;
            v16bf x0 = load_a_frag(h0b_cur, b0,      ks * 32, lane);
            v16bf x1 = load_a_frag(h0b_cur, b0 + 16, ks * 32, lane);
            v16bf y0 = load_a_frag(h1b_in,  b0,      ks * 32, lane);
            v16bf y1 = load_a_frag(h1b_in,  b0 + 16, ks * 32, lane);
            v16bf fiR = lds_frag(ldsW, ksl * 6 + 0, lane);
            v16bf fiZ = lds_frag(ldsW, ksl * 6 + 1, lane);
            v16bf fiN = lds_frag(ldsW, ksl * 6 + 2, lane);
            v16bf fhR = lds_frag(ldsW, ksl * 6 + 3, lane);
            v16bf fhZ = lds_frag(ldsW, ksl * 6 + 4, lane);
            v16bf fhN = lds_frag(ldsW, ksl * 6 + 5, lane);
            aR[0] = WMMA_BF16(x0, fiR, aR[0]);  aR[1] = WMMA_BF16(x1, fiR, aR[1]);
            aZ[0] = WMMA_BF16(x0, fiZ, aZ[0]);  aZ[1] = WMMA_BF16(x1, fiZ, aZ[1]);
            aN[0] = WMMA_BF16(x0, fiN, aN[0]);  aN[1] = WMMA_BF16(x1, fiN, aN[1]);
            hR[0] = WMMA_BF16(y0, fhR, hR[0]);  hR[1] = WMMA_BF16(y1, fhR, hR[1]);
            hZ[0] = WMMA_BF16(y0, fhZ, hZ[0]);  hZ[1] = WMMA_BF16(y1, fhZ, hZ[1]);
            hN[0] = WMMA_BF16(y0, fhN, hN[0]);  hN[1] = WMMA_BF16(y1, fhN, hN[1]);
        }
    }

    const int n = lane & 15, half = lane >> 4;
    const int j = j0 + n;
    const float bihr = b_ih1[j], bihz = b_ih1[HH + j], bihn = b_ih1[2 * HH + j];
    const float bhhr = b_hh1[j], bhhz = b_hh1[HH + j], bhhn = b_hh1[2 * HH + j];

#pragma unroll
    for (int tt = 0; tt < 2; ++tt) {
#pragma unroll
        for (int i = 0; i < 8; ++i) {
            int b = b0 + tt * 16 + half * 8 + i;
            float r  = sigmoidf_(aR[tt][i] + bihr + hR[tt][i] + bhhr);
            float z  = sigmoidf_(aZ[tt][i] + bihz + hZ[tt][i] + bhhz);
            float ng = tanhf(aN[tt][i] + bihn + r * (hN[tt][i] + bhhn));
            float hp = h1f_in[(size_t)b * HH + j];
            float hn = (1.0f - z) * ng + z * hp;
            h1f_out[(size_t)b * HH + j] = hn;
            h1b_out[(size_t)b * HH + j] = f2bf(hn);
        }
    }
}

// ---------------- linear head: out = h1 @ W_out^T + b_out ----------------
__global__ void gru_head_kernel(const float* __restrict__ h1,
                                const float* __restrict__ Wo,
                                const float* __restrict__ bo,
                                float* __restrict__ out) {
    int b = blockIdx.x * blockDim.x + threadIdx.x;
    if (b >= BB) return;
#pragma unroll
    for (int o = 0; o < OUTN; ++o) {
        float s = bo[o];
        for (int k = 0; k < HH; ++k) s += h1[(size_t)b * HH + k] * Wo[(size_t)o * HH + k];
        out[(size_t)b * OUTN + o] = s;
    }
}

extern "C" void kernel_launch(void* const* d_in, const int* in_sizes, int n_in,
                              void* d_out, int out_size, void* d_ws, size_t ws_size,
                              hipStream_t stream) {
    const float* x     = (const float*)d_in[0];
    const float* W_ih0 = (const float*)d_in[1];
    const float* W_hh0 = (const float*)d_in[2];
    const float* b_ih0 = (const float*)d_in[3];
    const float* b_hh0 = (const float*)d_in[4];
    const float* W_ih1 = (const float*)d_in[5];
    const float* W_hh1 = (const float*)d_in[6];
    const float* b_ih1 = (const float*)d_in[7];
    const float* b_hh1 = (const float*)d_in[8];
    const float* W_out = (const float*)d_in[9];
    const float* b_out = (const float*)d_in[10];

    const size_t BH = (size_t)BB * HH;       // 131072
    const size_t WE = (size_t)3 * HH * HH;   // 786432 elements per packed matrix
    // workspace (~7.5 MB): ping-pong h states + prepacked bf16 weights
    float*  h0f    = (float*)d_ws;                 // [2][B*H] fp32
    float*  h1f    = h0f + 2 * BH;                 // [2][B*H] fp32
    __bf16* h0b    = (__bf16*)(h1f + 2 * BH);      // [2][B*H] bf16
    __bf16* h1b    = h0b + 2 * BH;                 // [2][B*H] bf16
    __bf16* packW0 = h1b + 2 * BH;                 // [3H*H] packed W_hh0
    __bf16* packIH = packW0 + WE;                  // [3H*H] packed W_ih1
    __bf16* packHH = packIH + WE;                  // [3H*H] packed W_hh1

    gru_pack_kernel<<<192, 256, 0, stream>>>(W_hh0, packW0);
    gru_pack_kernel<<<192, 256, 0, stream>>>(W_ih1, packIH);
    gru_pack_kernel<<<192, 256, 0, stream>>>(W_hh1, packHH);
    gru_init_kernel<<<(int)((BH + 255) / 256), 256, 0, stream>>>(h0f, h1f, h0b, h1b);

    for (int t = 0; t < TT; ++t) {
        const int pi = t & 1, po = (t + 1) & 1;
        gru_layer0_kernel<<<HH / 16, 256, 0, stream>>>(
            x, t, W_ih0, packW0, b_ih0, b_hh0,
            h0f + pi * BH, h0b + pi * BH,
            h0f + po * BH, h0b + po * BH);
        gru_layer1_kernel<<<HH / 16, 256, 0, stream>>>(
            packIH, packHH, b_ih1, b_hh1,
            h0b + po * BH,                        // fresh layer-0 output
            h1f + pi * BH, h1b + pi * BH,
            h1f + po * BH, h1b + po * BH);
    }
    // after T=512 steps the last write parity is (511+1)&1 == 0
    gru_head_kernel<<<1, 256, 0, stream>>>(h1f + 0, W_out, b_out, (float*)d_out);
}
